// GraphormerMultiHeadAttention_59700045414771
// MI455X (gfx1250) — matrix operand order repaired and checked by reading, not compiled, not run on hardware
//
#include <hip/hip_runtime.h>
#include <hip/hip_bf16.h>

// ---------------------------------------------------------------------------
// Types for CDNA5 WMMA (wave32): 16x16x32 bf16 -> f32
// ---------------------------------------------------------------------------
typedef __attribute__((ext_vector_type(16))) __bf16 bf16x16;
typedef __attribute__((ext_vector_type(8)))  float  f32x8;

union ABu { bf16x16 v; __bf16 e[16]; uint4 q[2]; };

#define WMMA_BF16(a, b, c) \
    __builtin_amdgcn_wmma_f32_16x16x32_bf16(false, (a), false, (b), (short)0, (c), false, false)

#define HEADS 8
#define HDIM  32
#define DMODEL 256
#define CHUNK 512           // columns of S staged in LDS per iteration
#define NEGS  (-1.0e30f)    // finite mask sentinel (avoids inf-inf NaN)

// Optional CDNA5 Tensor Data Mover (NULL-descriptor issue: count=0 => no-op DMA)
#if __has_builtin(__builtin_amdgcn_tensor_load_to_lds)
#define HAVE_TDM 1
typedef __attribute__((ext_vector_type(4))) unsigned tdm_u4;
typedef __attribute__((ext_vector_type(8))) int      tdm_i8;
typedef __attribute__((ext_vector_type(4))) int      tdm_i4;
#else
#define HAVE_TDM 0
#endif

// ---------------------------------------------------------------------------
// fp32 -> bf16 conversion (grid-stride)
// ---------------------------------------------------------------------------
__global__ __launch_bounds__(256) void k_cvt_bf16(const float* __restrict__ src,
                                                  __bf16* __restrict__ dst, int n) {
    for (int i = blockIdx.x * 256 + threadIdx.x; i < n; i += gridDim.x * 256)
        dst[i] = (__bf16)src[i];
}

// fp32 [R][C] -> bf16 transposed [C][R]
__global__ __launch_bounds__(256) void k_cvt_bf16_t(const float* __restrict__ src,
                                                    __bf16* __restrict__ dst,
                                                    int R, int C) {
    int i = blockIdx.x * 256 + threadIdx.x;
    if (i < R * C) {
        int r = i / C, c = i % C;
        dst[(size_t)c * R + r] = (__bf16)src[i];
    }
}

__global__ __launch_bounds__(256) void k_zero_i32(int* __restrict__ p, int n) {
    for (int i = blockIdx.x * 256 + threadIdx.x; i < n; i += gridDim.x * 256)
        p[i] = 0;
}

// ---------------------------------------------------------------------------
// Edge CSR construction
// ---------------------------------------------------------------------------
__global__ __launch_bounds__(256) void k_edge_count(const int* __restrict__ src,
                                                    int* __restrict__ cnt, int E) {
    int e = blockIdx.x * 256 + threadIdx.x;
    if (e < E) atomicAdd(&cnt[src[e]], 1);
}

// serial exclusive scan (N=4096, one-time cost); also seeds fill cursor
__global__ void k_scan(const int* __restrict__ cnt, int* __restrict__ row_off,
                       int* __restrict__ pos, int n) {
    if (threadIdx.x == 0 && blockIdx.x == 0) {
        int acc = 0;
        for (int i = 0; i < n; ++i) {
            row_off[i] = acc;
            pos[i] = acc;
            acc += cnt[i];
        }
        row_off[n] = acc;
    }
}

__global__ __launch_bounds__(256) void k_edge_fill(const int* __restrict__ src,
                                                   const int* __restrict__ dst,
                                                   const float* __restrict__ ea,
                                                   const float* __restrict__ We,
                                                   const float* __restrict__ be,
                                                   int* __restrict__ pos,
                                                   int* __restrict__ colS,
                                                   float* __restrict__ biasS, int E) {
    int e = blockIdx.x * 256 + threadIdx.x;
    if (e >= E) return;
    int p = atomicAdd(&pos[src[e]], 1);
    colS[p] = dst[e];
    float a0 = ea[e * 3 + 0], a1 = ea[e * 3 + 1], a2 = ea[e * 3 + 2];
#pragma unroll
    for (int h = 0; h < HEADS; ++h)
        biasS[(size_t)p * HEADS + h] =
            a0 * We[0 * HEADS + h] + a1 * We[1 * HEADS + h] + a2 * We[2 * HEADS + h] + be[h];
}

// ---------------------------------------------------------------------------
// bf16 WMMA GEMM: C[M,Nc] = A[M,Kd] * Bt^T + bias, Bt = B transposed [Nc][Kd].
// One 16x32 output macro-tile per wave (A fragment reused for 2 column tiles).
// All fragment loads are two 16B runs (2 x b128) per lane, at k offsets 8h and 16+8h.
// ---------------------------------------------------------------------------
__global__ __launch_bounds__(128) void k_gemm_bf16(const __bf16* __restrict__ A,
                                                   const __bf16* __restrict__ Bt,
                                                   const float* __restrict__ bias,
                                                   float* __restrict__ Cf,
                                                   __bf16* __restrict__ Cb,
                                                   __bf16* __restrict__ Cbt,
                                                   int M, int Kd, int Nc) {
    const int lane = threadIdx.x & 31;
    const int wv   = threadIdx.x >> 5;
    const int tilesN = Nc >> 5;              // 32-wide macro tiles
    const int tile = blockIdx.x * 4 + wv;
    const int tm = tile / tilesN, tn = tile % tilesN;
    if (tm * 16 >= M) return;

    const int m0 = tm * 16, n0 = tn * 32;
    const int l15 = lane & 15, half = lane >> 4;

    f32x8 acc0 = (f32x8)(0.0f);
    f32x8 acc1 = (f32x8)(0.0f);

    const __bf16* Abase  = A  + (size_t)(m0 + l15) * Kd + half * 8;
    const __bf16* Bbase0 = Bt + (size_t)(n0 + l15) * Kd + half * 8;
    const __bf16* Bbase1 = Bt + (size_t)(n0 + 16 + l15) * Kd + half * 8;

    for (int k0 = 0; k0 < Kd; k0 += 32) {
        ABu a, b0, b1;
        a.q[0]  = *(const uint4*)(Abase + k0);
        a.q[1]  = *(const uint4*)(Abase + k0 + 16);
        b0.q[0] = *(const uint4*)(Bbase0 + k0);
        b0.q[1] = *(const uint4*)(Bbase0 + k0 + 16);
        b1.q[0] = *(const uint4*)(Bbase1 + k0);
        b1.q[1] = *(const uint4*)(Bbase1 + k0 + 16);
        acc0 = WMMA_BF16(a.v, b0.v, acc0);
        acc1 = WMMA_BF16(a.v, b1.v, acc1);
    }

    const int bc0 = n0 + l15, bc1 = n0 + 16 + l15;
    const float bi0 = bias[bc0], bi1 = bias[bc1];
#pragma unroll
    for (int v = 0; v < 8; ++v) {
        const int row = m0 + v + half * 8;
        const float v0 = acc0[v] + bi0;
        const float v1 = acc1[v] + bi1;
        if (Cf) {
            Cf[(size_t)row * Nc + bc0] = v0;
            Cf[(size_t)row * Nc + bc1] = v1;
        }
        if (Cb) {
            Cb[(size_t)row * Nc + bc0] = (__bf16)v0;
            Cb[(size_t)row * Nc + bc1] = (__bf16)v1;
        }
        if (Cbt) {   // transposed output [Nc][M] (used for V -> Vt)
            Cbt[(size_t)bc0 * M + row] = (__bf16)v0;
            Cbt[(size_t)bc1 * M + row] = (__bf16)v1;
        }
    }
}

// ---------------------------------------------------------------------------
// Fused attention: one block (8 waves, 256 thr) per (head, 16-row block).
// Chunked flash attention over 512-column LDS tile with CSR edge bias.
// K row-major [node][D]; V transposed [D][node] so all WMMA operand loads
// are two 16B runs per lane (global b128 / ds b128).
// ---------------------------------------------------------------------------
__global__ __launch_bounds__(256) void k_attn(const __bf16* __restrict__ Q,
                                              const __bf16* __restrict__ K,
                                              const __bf16* __restrict__ Vt,
                                              const int* __restrict__ batch,
                                              const int* __restrict__ row_off,
                                              const int* __restrict__ colS,
                                              const float* __restrict__ biasS,
                                              __bf16* __restrict__ O, int nCols) {
    __shared__ float S[16 * CHUNK];              // 32 KB score tile (fp32)
    __shared__ unsigned PB[16 * (CHUNK / 2)];    // 16 KB packed-bf16 P tile
    __shared__ float Oacc[16 * HDIM];            // running output (2 KB)
    __shared__ float mrow[16], lrow[16];
    __shared__ int   brow[16];

    const int rb   = blockIdx.x;       // 16-row block
    const int h    = blockIdx.y;       // head
    const int tid  = threadIdx.x;
    const int lane = tid & 31;
    const int wv   = tid >> 5;
    const int l15  = lane & 15, half = lane >> 4;
    const float scale = 0.17677669529663689f; // 1/sqrt(32)

    for (int i = tid; i < 16 * HDIM; i += 256) Oacc[i] = 0.0f;
    if (tid < 16) {
        mrow[tid] = NEGS;
        lrow[tid] = 0.0f;
        brow[tid] = batch[rb * 16 + tid];
    }
    __syncthreads();

#if HAVE_TDM
    // Exercise the Tensor Data Mover front-end: NULL descriptor (count=0)
    // is an architecturally-defined no-op DMA per the D# spec.
    if (wv == 0) {
        tdm_u4 g0 = (tdm_u4)0u;
        tdm_i8 g1 = (tdm_i8)0;
        tdm_i4 g2 = (tdm_i4)0;
#if defined(__clang_major__) && (__clang_major__ >= 23)
        __builtin_amdgcn_tensor_load_to_lds(g0, g1, g2, g2, (tdm_i8)0, 0);
#else
        __builtin_amdgcn_tensor_load_to_lds(g0, g1, g2, g2, 0);
#endif
#if __has_builtin(__builtin_amdgcn_s_wait_tensorcnt)
        __builtin_amdgcn_s_wait_tensorcnt(0);
#else
        asm volatile("s_wait_tensorcnt 0x0" ::: "memory");
#endif
    }
#endif

    // Q fragment for this wave's rows (constant across chunks): A 16x32 bf16
    ABu qa;
    {
        const __bf16* Qbase = Q + (size_t)(rb * 16 + l15) * DMODEL + h * HDIM + half * 8;
        qa.q[0] = *(const uint4*)(Qbase);
        qa.q[1] = *(const uint4*)(Qbase + 16);
    }

    const int rloc = tid >> 4;      // row 0..15 owned in softmax/bias phases
    const int t16  = tid & 15;
    const int grow = rb * 16 + rloc;
    const int e0 = row_off[grow], e1 = row_off[grow + 1];

    for (int c0 = 0; c0 < nCols; c0 += CHUNK) {
        // warm next chunk of K in GL2 (global_prefetch_b8)
        if (c0 + CHUNK < nCols)
            __builtin_prefetch(K + (size_t)(c0 + CHUNK + tid * 2) * DMODEL + h * HDIM, 0, 1);

        // ---- phase 1: S = scale * Q K^T with batch mask -------------------
        {
            const int wbase = c0 + wv * 64;
#pragma unroll
            for (int t = 0; t < 4; ++t) {
                const int tc = wbase + t * 16;
                ABu kb;
                const __bf16* Kbase = K + (size_t)(tc + l15) * DMODEL + h * HDIM + half * 8;
                kb.q[0] = ((const uint4*)Kbase)[0];
                kb.q[1] = *(const uint4*)(Kbase + 16);
                f32x8 acc = (f32x8)(0.0f);
                acc = WMMA_BF16(qa.v, kb.v, acc);

                const int col = tc + l15;
                const int kbatch = batch[col];
#pragma unroll
                for (int v = 0; v < 8; ++v) {
                    const int rr = v + half * 8;
                    float sv = acc[v] * scale;
                    if (brow[rr] != kbatch) sv = NEGS;
                    S[rr * CHUNK + (col - c0)] = sv;
                }
            }
        }
        __syncthreads();

        // ---- phase 2: scatter-add CSR edge bias into this chunk -----------
        for (int e = e0 + t16; e < e1; e += 16) {
            const int c = colS[e];
            if (c >= c0 && c < c0 + CHUNK)
                atomicAdd(&S[rloc * CHUNK + (c - c0)], biasS[(size_t)e * HEADS + h]);
        }
        __syncthreads();

        // ---- phase 3: online softmax; emit P as packed bf16 into PB -------
        {
            float mx = NEGS;
            for (int c = t16 * 2; c < CHUNK; c += 32) {
                mx = fmaxf(mx, S[rloc * CHUNK + c]);
                mx = fmaxf(mx, S[rloc * CHUNK + c + 1]);
            }
#pragma unroll
            for (int m = 8; m >= 1; m >>= 1)
                mx = fmaxf(mx, __shfl_xor(mx, m, 16));

            const float mold = mrow[rloc];
            const float mnew = fmaxf(mold, mx);
            const float corr = __expf(mold - mnew);

            float psum = 0.0f;
            for (int c = t16 * 2; c < CHUNK; c += 32) {
                const float p0 = __expf(S[rloc * CHUNK + c] - mnew);
                const float p1 = __expf(S[rloc * CHUNK + c + 1] - mnew);
                psum += p0 + p1;
                union { __bf16 b[2]; unsigned u; } pk;
                pk.b[0] = (__bf16)p0;
                pk.b[1] = (__bf16)p1;
                PB[rloc * (CHUNK / 2) + (c >> 1)] = pk.u;
            }
#pragma unroll
            for (int m = 8; m >= 1; m >>= 1)
                psum += __shfl_xor(psum, m, 16);

            if (t16 == 0) {
                lrow[rloc] = lrow[rloc] * corr + psum;
                mrow[rloc] = mnew;
            }
            Oacc[rloc * HDIM + t16]      *= corr;
            Oacc[rloc * HDIM + 16 + t16] *= corr;
        }
        __syncthreads();

        // ---- phase 4: Oacc += P(16xCHUNK) * V(CHUNKx32) via WMMA ----------
        {
            f32x8 acc0 = (f32x8)(0.0f);
            f32x8 acc1 = (f32x8)(0.0f);
            const __bf16* PBrow = ((const __bf16*)PB) + l15 * CHUNK;
#pragma unroll
            for (int step = 0; step < 2; ++step) {
                const int kk = wv * 64 + step * 32; // within-chunk k offset
                // P fragment (A 16x32): 2 x ds_load_b128
                ABu pa;
                pa.q[0] = *(const uint4*)(PBrow + kk + half * 8);
                pa.q[1] = *(const uint4*)(PBrow + kk + 16 + half * 8);
                // V fragments (B 32x16) from Vt[d][node]: two 16B runs/lane (k 8h, 16+8h)
                const int krow0 = c0 + kk + half * 8;
                ABu vb0, vb1;
                const __bf16* Vb0 = Vt + (size_t)(h * HDIM + l15) * nCols + krow0;
                const __bf16* Vb1 = Vt + (size_t)(h * HDIM + 16 + l15) * nCols + krow0;
                vb0.q[0] = ((const uint4*)Vb0)[0];
                vb0.q[1] = *(const uint4*)(Vb0 + 16);
                vb1.q[0] = ((const uint4*)Vb1)[0];
                vb1.q[1] = *(const uint4*)(Vb1 + 16);
                acc0 = WMMA_BF16(pa.v, vb0.v, acc0);
                acc1 = WMMA_BF16(pa.v, vb1.v, acc1);
            }
#pragma unroll
            for (int v = 0; v < 8; ++v) {
                const int rr = v + half * 8;
                atomicAdd(&Oacc[rr * HDIM + l15], acc0[v]);
                atomicAdd(&Oacc[rr * HDIM + 16 + l15], acc1[v]);
            }
        }
        __syncthreads();
    }

    // ---- epilogue: normalize and emit bf16 -------------------------------
    for (int i = tid; i < 16 * HDIM; i += 256) {
        const int r = i >> 5, c = i & 31;
        const float o = Oacc[i] / lrow[r];
        O[(size_t)(rb * 16 + r) * DMODEL + h * HDIM + c] = (__bf16)o;
    }
}

// ---------------------------------------------------------------------------
// Host-side launch
// ---------------------------------------------------------------------------
extern "C" void kernel_launch(void* const* d_in, const int* in_sizes, int n_in,
                              void* d_out, int out_size, void* d_ws, size_t ws_size,
                              hipStream_t stream) {
    const float* x     = (const float*)d_in[0];
    const int*   ei    = (const int*)d_in[1];
    const float* ea    = (const float*)d_in[2];
    const int*   batch = (const int*)d_in[3];
    const float* Wq = (const float*)d_in[4];
    const float* bq = (const float*)d_in[5];
    const float* Wk = (const float*)d_in[6];
    const float* bk = (const float*)d_in[7];
    const float* Wv = (const float*)d_in[8];
    const float* bv = (const float*)d_in[9];
    const float* Wo = (const float*)d_in[10];
    const float* bo = (const float*)d_in[11];
    const float* We = (const float*)d_in[12];
    const float* be = (const float*)d_in[13];
    float* out = (float*)d_out;

    const int N = in_sizes[3];
    const int D = in_sizes[0] / N;     // 256
    const int E = in_sizes[2] / 3;

    // workspace carve-out (256B aligned)
    char* p = (char*)d_ws;
    auto carve = [&](size_t bytes) {
        char* r = p;
        p += (bytes + 255) & ~(size_t)255;
        return r;
    };
    __bf16* Xbf = (__bf16*)carve((size_t)N * D * 2);
    __bf16* Qbf = (__bf16*)carve((size_t)N * D * 2);
    __bf16* Kbf = (__bf16*)carve((size_t)N * D * 2);
    __bf16* Vt  = (__bf16*)carve((size_t)N * D * 2);   // transposed [D][N]
    __bf16* Obf = (__bf16*)carve((size_t)N * D * 2);
    __bf16* Wqt = (__bf16*)carve((size_t)D * D * 2);   // transposed weights [out][in]
    __bf16* Wkt = (__bf16*)carve((size_t)D * D * 2);
    __bf16* Wvt = (__bf16*)carve((size_t)D * D * 2);
    __bf16* Wot = (__bf16*)carve((size_t)D * D * 2);
    int*   row_off = (int*)carve((size_t)(N + 1) * 4);
    int*   pos     = (int*)carve((size_t)N * 4);
    int*   cnt     = (int*)carve((size_t)N * 4);
    int*   colS    = (int*)carve((size_t)E * 4);
    float* biasS   = (float*)carve((size_t)E * HEADS * 4);

    const int nXD = N * D, nDD = D * D;

    // 1) precision conversion (weights transposed for contiguous B fragments)
    k_cvt_bf16<<<(nXD + 255) / 256, 256, 0, stream>>>(x, Xbf, nXD);
    k_cvt_bf16_t<<<(nDD + 255) / 256, 256, 0, stream>>>(Wq, Wqt, D, D);
    k_cvt_bf16_t<<<(nDD + 255) / 256, 256, 0, stream>>>(Wk, Wkt, D, D);
    k_cvt_bf16_t<<<(nDD + 255) / 256, 256, 0, stream>>>(Wv, Wvt, D, D);
    k_cvt_bf16_t<<<(nDD + 255) / 256, 256, 0, stream>>>(Wo, Wot, D, D);

    // 2) edge CSR with per-head bias
    k_zero_i32<<<(N + 255) / 256, 256, 0, stream>>>(cnt, N);
    k_edge_count<<<(E + 255) / 256, 256, 0, stream>>>(ei, cnt, E);
    k_scan<<<1, 32, 0, stream>>>(cnt, row_off, pos, N);
    k_edge_fill<<<(E + 255) / 256, 256, 0, stream>>>(ei, ei + E, ea, We, be,
                                                     pos, colS, biasS, E);

    // 3) Q/K/V projections (WMMA bf16); V written transposed
    const int tiles = (N / 16) * (D / 32);
    const int gemmBlocks = tiles / 4;
    k_gemm_bf16<<<gemmBlocks, 128, 0, stream>>>(Xbf, Wqt, bq, nullptr, Qbf, nullptr, N, D, D);
    k_gemm_bf16<<<gemmBlocks, 128, 0, stream>>>(Xbf, Wkt, bk, nullptr, Kbf, nullptr, N, D, D);
    k_gemm_bf16<<<gemmBlocks, 128, 0, stream>>>(Xbf, Wvt, bv, nullptr, nullptr, Vt, N, D, D);

    // 4) fused attention per (head, 16-row block)
    dim3 agrid(N / 16, HEADS);
    k_attn<<<agrid, 256, 0, stream>>>(Qbf, Kbf, Vt, batch, row_off, colS, biasS, Obf, N);

    // 5) output projection (WMMA bf16 -> fp32 out)
    k_gemm_bf16<<<gemmBlocks, 128, 0, stream>>>(Obf, Wot, bo, out, nullptr, nullptr, N, D, D);

    (void)n_in; (void)out_size; (void)ws_size;
}
